// Mamba2Block_43963285242724
// MI455X (gfx1250) — compile-verified, matrix operand order checked by reading, not run on hardware
//
#include <hip/hip_runtime.h>
#include <hip/hip_bf16.h>
#include <cstdint>
#include <cstddef>

// ---------------- problem constants (from reference) ----------------
#define BATCH   2
#define SEQ     2048
#define HIDDEN  2048
#define NH      64
#define PP      64               /* P head dim */
#define II      4096             /* I = NH*P   */
#define NN      128              /* state N    */
#define KC      4                /* conv K     */
#define CONV_DIM (II + 2*NN)     /* 4352 */
#define PROJ   (II + CONV_DIM + NH) /* 8512 */
#define BS     (BATCH*SEQ)       /* 4096 */
#define EPSV   1e-6f

// enable the gfx1250 async global->LDS staging path (ASYNCcnt-tracked)
#define GEMM_ASYNC_STAGE 1

typedef __attribute__((ext_vector_type(16))) __bf16 v16bf;
typedef __attribute__((ext_vector_type(8)))  __bf16 v8bf;
typedef __attribute__((ext_vector_type(8)))  float  v8f;

// ---------------- helpers ----------------
__device__ __forceinline__ __bf16 f32_to_bf16(float f) {
  uint32_t u = __builtin_bit_cast(uint32_t, f);
  u += 0x7fffu + ((u >> 16) & 1u);          // round-to-nearest-even
  uint16_t h = (uint16_t)(u >> 16);
  return __builtin_bit_cast(__bf16, h);
}
__device__ __forceinline__ float siluf(float x) {
  return x / (1.0f + __expf(-x));
}
__device__ __forceinline__ float softplusf(float x) {
  return (x > 20.0f) ? x : log1pf(__expf(x));
}

#if GEMM_ASYNC_STAGE
typedef const __attribute__((address_space(3))) char* lds_cptr_t;
__device__ __forceinline__ uint32_t lds_off_of(const void* p) {
  return (uint32_t)(uintptr_t)(lds_cptr_t)p;   // 32-bit LDS byte offset
}
// GLOBAL_LOAD_ASYNC_TO_LDS_B128 (GV mode): VDST = LDS byte addr, VADDR = 64b global
__device__ __forceinline__ void async_g2l_b128(uint32_t lds_byte, const void* gptr) {
  asm volatile("global_load_async_to_lds_b128 %0, %1, off"
               :: "v"(lds_byte), "v"(gptr) : "memory");
}
__device__ __forceinline__ void wait_async0() {
  asm volatile("s_wait_asynccnt 0x0" ::: "memory");
}
#endif

// ---------------- 1) fp32 -> bf16 conversion (bandwidth bound) ----------------
__global__ void cvt_f32_bf16_k(const float* __restrict__ src,
                               __bf16* __restrict__ dst, long n) {
  long i = (long)(blockIdx.x) * blockDim.x + threadIdx.x;
  long stride = (long)gridDim.x * blockDim.x;
  for (long j = i * 4; j < n; j += stride * 4) {
    float4 v = *(const float4*)(src + j);
    dst[j + 0] = f32_to_bf16(v.x);
    dst[j + 1] = f32_to_bf16(v.y);
    dst[j + 2] = f32_to_bf16(v.z);
    dst[j + 3] = f32_to_bf16(v.w);
  }
}

// ---------------- 2) BF16 WMMA GEMM: C[M,N] = A[M,K] * W[N,K]^T ----------------
// 256 threads (8 waves, 2x4), block tile 128(M) x 128(N), K-step 64,
// double-buffered LDS filled with async global->LDS loads.
#define LDSTR 72   /* padded bf16 row stride for a 64-wide K slice (144 B) */

__global__ __launch_bounds__(256)
void gemm_bf16_wmma_k(const __bf16* __restrict__ A,
                      const __bf16* __restrict__ W,
                      float* __restrict__ C,
                      int M, int N, int K) {
  __shared__ __bf16 As[2][128 * LDSTR];
  __shared__ __bf16 Ws[2][128 * LDSTR];

  const int tid   = threadIdx.x;
  const int lane  = tid & 31;
  const int wid   = tid >> 5;
  const int waveM = wid >> 2;      // 0..1
  const int waveN = wid & 3;       // 0..3
  const int m0    = blockIdx.y * 128;
  const int n0    = blockIdx.x * 128;
  const int half  = lane >> 4;     // K-half selector (WMMA bf16 A/B layout)
  const int r     = lane & 15;

  v8f acc[4][2];
  #pragma unroll
  for (int mt = 0; mt < 4; ++mt)
    #pragma unroll
    for (int nt = 0; nt < 2; ++nt)
      acc[mt][nt] = (v8f){};

  // stage one 128x64 bf16 K-slice of A and W into LDS buffer `buf`
  auto stage = [&](int buf, int kk) {
    #pragma unroll
    for (int c = 0; c < 4; ++c) {
      int chunk = tid + c * 256;          // 0..1023
      int row   = chunk >> 3;             // 0..127
      int kq    = (chunk & 7) * 8;        // 0,8,..,56 halfs
#if GEMM_ASYNC_STAGE
      async_g2l_b128(lds_off_of(&As[buf][row * LDSTR + kq]),
                     A + (size_t)(m0 + row) * K + kk + kq);
      int rw = n0 + row;
      if (rw < N) {
        async_g2l_b128(lds_off_of(&Ws[buf][row * LDSTR + kq]),
                       W + (size_t)rw * K + kk + kq);
      } else {
        *(v8bf*)&Ws[buf][row * LDSTR + kq] = (v8bf){};
      }
#else
      *(v8bf*)&As[buf][row * LDSTR + kq] =
          *(const v8bf*)(A + (size_t)(m0 + row) * K + kk + kq);
      int rw = n0 + row;
      v8bf wv = (v8bf){};
      if (rw < N) wv = *(const v8bf*)(W + (size_t)rw * K + kk + kq);
      *(v8bf*)&Ws[buf][row * LDSTR + kq] = wv;
#endif
    }
  };

  // prologue: fill buffer 0
  stage(0, 0);
#if GEMM_ASYNC_STAGE
  wait_async0();
#endif
  __syncthreads();

  const int T = K >> 6;                   // K-steps of 64
  for (int kt = 0; kt < T; ++kt) {
    const int cur = kt & 1;
    // kick off the next tile's fill while we compute on the current one
    if (kt + 1 < T) stage(1 - cur, (kt + 1) << 6);

    // 2 sub-slices of K=32 -> 16 WMMAs per iteration
    #pragma unroll
    for (int ks = 0; ks < 2; ++ks) {
      v16bf afr[4], bfr[2];
      #pragma unroll
      for (int mt = 0; mt < 4; ++mt) {
        int base = (waveM * 64 + mt * 16 + r) * LDSTR + ks * 32 + half * 8;
        v8bf lo = *(const v8bf*)&As[cur][base];
        v8bf hi = *(const v8bf*)&As[cur][base + 16];
        afr[mt] = __builtin_shufflevector(lo, hi,
                   0,1,2,3,4,5,6,7,8,9,10,11,12,13,14,15);
      }
      #pragma unroll
      for (int nt = 0; nt < 2; ++nt) {
        int base = (waveN * 32 + nt * 16 + r) * LDSTR + ks * 32 + half * 8;
        v8bf lo = *(const v8bf*)&Ws[cur][base];
        v8bf hi = *(const v8bf*)&Ws[cur][base + 16];
        bfr[nt] = __builtin_shufflevector(lo, hi,
                   0,1,2,3,4,5,6,7,8,9,10,11,12,13,14,15);
      }
      #pragma unroll
      for (int mt = 0; mt < 4; ++mt)
        #pragma unroll
        for (int nt = 0; nt < 2; ++nt)
          acc[mt][nt] = __builtin_amdgcn_wmma_f32_16x16x32_bf16(
              false, afr[mt], false, bfr[nt], (short)0, acc[mt][nt],
              false, false);
    }

    if (kt + 1 < T) {
#if GEMM_ASYNC_STAGE
      wait_async0();                      // own wave's fills landed in LDS
#endif
      __syncthreads();                    // everyone's fills visible / reads done
    }
  }

  // --- epilogue: D layout: VGPR rr, lane L -> row = rr + 8*(L>>4), col = L&15
  #pragma unroll
  for (int mt = 0; mt < 4; ++mt) {
    #pragma unroll
    for (int nt = 0; nt < 2; ++nt) {
      int col = n0 + waveN * 32 + nt * 16 + r;
      if (col < N) {
        #pragma unroll
        for (int rr = 0; rr < 8; ++rr) {
          int row = m0 + waveM * 64 + mt * 16 + half * 8 + rr;
          C[(size_t)row * N + col] = acc[mt][nt][rr];
        }
      }
    }
  }
}

// ---------------- 3) depthwise conv (K=4) + SiLU ----------------
__global__ void conv_silu_k(const float* __restrict__ proj,
                            const unsigned char* __restrict__ mask,
                            const float* __restrict__ conv_w,
                            const float* __restrict__ conv_b,
                            float* __restrict__ convout) {
  long idx = (long)blockIdx.x * blockDim.x + threadIdx.x;
  long total = (long)BS * CONV_DIM;
  if (idx >= total) return;
  int c  = (int)(idx % CONV_DIM);
  long bs = idx / CONV_DIM;
  int s  = (int)(bs % SEQ);
  int b  = (int)(bs / SEQ);
  float acc = conv_b[c];
  #pragma unroll
  for (int k = 0; k < KC; ++k) {
    int sp = s - (KC - 1) + k;
    if (sp >= 0) {
      float m = (float)mask[(size_t)b * SEQ + sp];
      float v = proj[((size_t)b * SEQ + sp) * PROJ + II + c] * m;
      acc = fmaf(v, conv_w[c * KC + k], acc);
    }
  }
  convout[idx] = siluf(acc);
}

// ---------------- 4) dt = softplus(raw+bias)*mask ; dA = exp(dt*A) ----------------
__global__ void dt_k(const float* __restrict__ proj,
                     const unsigned char* __restrict__ mask,
                     const float* __restrict__ dt_bias,
                     const float* __restrict__ A_log,
                     float* __restrict__ dt,
                     float* __restrict__ dA) {
  long idx = (long)blockIdx.x * blockDim.x + threadIdx.x;
  if (idx >= (long)BS * NH) return;
  int h  = (int)(idx % NH);
  long bs = idx / NH;
  float raw = proj[bs * PROJ + II + CONV_DIM + h];
  float d = softplusf(raw + dt_bias[h]);
  d = fminf(fmaxf(d, 0.0f), 1e9f);
  d *= (float)mask[bs];
  float A = -__expf(A_log[h]);
  dt[idx] = d;
  dA[idx] = __expf(d * A);
}

// ---------------- 5) sequential SSM scan: one block per (b,h) ----------------
// 256 threads; thread = p*4 + ng owns state[p, ng*32 .. ng*32+31] in registers.
__global__ __launch_bounds__(256)
void ssm_scan_k(const float* __restrict__ convout,
                const float* __restrict__ dt,
                const float* __restrict__ dA,
                const float* __restrict__ Dv,
                float* __restrict__ yss) {
  const int b = blockIdx.x >> 6;       // 0..1
  const int h = blockIdx.x & 63;       // 0..63
  const int tid = threadIdx.x;
  const int p  = tid >> 2;             // 0..63
  const int ng = tid & 3;              // 0..3
  const int n0 = ng * 32;
  const float Dh = Dv[h];

  __shared__ float Bs[NN];
  __shared__ float Cs[NN];
  __shared__ float xs[PP];

  float st[32];
  #pragma unroll
  for (int j = 0; j < 32; ++j) st[j] = 0.0f;

  for (int t = 0; t < SEQ; ++t) {
    const size_t rbase = ((size_t)b * SEQ + t) * CONV_DIM;
    __syncthreads();
    if (tid < 128) Bs[tid]       = convout[rbase + II + tid];
    else           Cs[tid - 128] = convout[rbase + II + NN + (tid - 128)];
    if (tid < PP)  xs[tid]       = convout[rbase + h * PP + tid];
    __syncthreads();

    const size_t sidx = ((size_t)b * SEQ + t) * NH + h;
    const float dAv = dA[sidx];
    const float dtv = dt[sidx];
    const float xv  = xs[p];
    const float coef = dtv * xv;

    float acc = 0.0f;
    #pragma unroll
    for (int j = 0; j < 32; ++j) {
      st[j] = fmaf(st[j], dAv, coef * Bs[n0 + j]);
      acc   = fmaf(st[j], Cs[n0 + j], acc);
    }
    // reduce over the 4 n-groups (lanes p*4..p*4+3, contiguous in a wave32)
    acc += __shfl_xor(acc, 1, 32);
    acc += __shfl_xor(acc, 2, 32);
    if (ng == 0)
      yss[(((size_t)b * SEQ + t) * NH + h) * PP + p] = acc + xv * Dh;
  }
}

// ---------------- 6) gate*SiLU + RMSNorm -> bf16 ----------------
__global__ __launch_bounds__(256)
void gate_norm_k(const float* __restrict__ yss,
                 const float* __restrict__ proj,
                 const float* __restrict__ norm_w,
                 __bf16* __restrict__ ybf) {
  const int row = blockIdx.x;           // 0..BS-1
  const int tid = threadIdx.x;
  __shared__ float red[256];

  float v[16];
  float ssq = 0.0f;
  #pragma unroll
  for (int j = 0; j < 16; ++j) {
    int i = tid + j * 256;
    float g = proj[(size_t)row * PROJ + i];
    float y = yss[(size_t)row * II + i] * siluf(g);
    v[j] = y;
    ssq = fmaf(y, y, ssq);
  }
  red[tid] = ssq;
  __syncthreads();
  for (int s = 128; s > 0; s >>= 1) {
    if (tid < s) red[tid] += red[tid + s];
    __syncthreads();
  }
  float scale = rsqrtf(red[0] / (float)II + EPSV);
  #pragma unroll
  for (int j = 0; j < 16; ++j) {
    int i = tid + j * 256;
    ybf[(size_t)row * II + i] = f32_to_bf16(v[j] * scale * norm_w[i]);
  }
}

// ---------------- launch ----------------
extern "C" void kernel_launch(void* const* d_in, const int* in_sizes, int n_in,
                              void* d_out, int out_size, void* d_ws, size_t ws_size,
                              hipStream_t stream) {
  const float*         hs       = (const float*)d_in[0];
  const unsigned char* mask     = (const unsigned char*)d_in[1];
  const float*         in_w     = (const float*)d_in[2];
  const float*         conv_w   = (const float*)d_in[3];
  const float*         conv_b   = (const float*)d_in[4];
  const float*         dt_bias  = (const float*)d_in[5];
  const float*         A_log    = (const float*)d_in[6];
  const float*         Dv       = (const float*)d_in[7];
  const float*         norm_w   = (const float*)d_in[8];
  const float*         out_w    = (const float*)d_in[9];
  float*               out      = (float*)d_out;

  // workspace carve-up
  char* ws = (char*)d_ws;
  size_t off = 0;
  auto carve = [&](size_t bytes) {
    size_t o = off; off = (off + bytes + 255) & ~(size_t)255; return o;
  };
  __bf16* Xbf    = (__bf16*)(ws + carve((size_t)BS * HIDDEN * 2));
  __bf16* Winbf  = (__bf16*)(ws + carve((size_t)PROJ * HIDDEN * 2));
  __bf16* Woutbf = (__bf16*)(ws + carve((size_t)HIDDEN * II * 2));
  float*  projb  = (float*)(ws + carve((size_t)BS * PROJ * 4));
  float*  convo  = (float*)(ws + carve((size_t)BS * CONV_DIM * 4));
  float*  dtb    = (float*)(ws + carve((size_t)BS * NH * 4));
  float*  dAb    = (float*)(ws + carve((size_t)BS * NH * 4));
  float*  yss    = (float*)(ws + carve((size_t)BS * II * 4));
  __bf16* ybf    = (__bf16*)(ws + carve((size_t)BS * II * 2));

  // 1) down-convert activations + weights to bf16
  {
    long n1 = (long)BS * HIDDEN;
    cvt_f32_bf16_k<<<(unsigned)((n1/4 + 255)/256), 256, 0, stream>>>(hs, Xbf, n1);
    long n2 = (long)PROJ * HIDDEN;
    cvt_f32_bf16_k<<<(unsigned)((n2/4 + 255)/256), 256, 0, stream>>>(in_w, Winbf, n2);
    long n3 = (long)HIDDEN * II;
    cvt_f32_bf16_k<<<(unsigned)((n3/4 + 255)/256), 256, 0, stream>>>(out_w, Woutbf, n3);
  }

  // 2) in_proj GEMM: [BS,HIDDEN] x [PROJ,HIDDEN]^T -> [BS,PROJ]
  {
    dim3 grid((PROJ + 127) / 128, BS / 128);
    gemm_bf16_wmma_k<<<grid, 256, 0, stream>>>(Xbf, Winbf, projb,
                                               BS, PROJ, HIDDEN);
  }

  // 3) conv + silu
  {
    long total = (long)BS * CONV_DIM;
    conv_silu_k<<<(unsigned)((total + 255)/256), 256, 0, stream>>>(
        projb, mask, conv_w, conv_b, convo);
  }

  // 4) dt / dA
  {
    long total = (long)BS * NH;
    dt_k<<<(unsigned)((total + 255)/256), 256, 0, stream>>>(
        projb, mask, dt_bias, A_log, dtb, dAb);
  }

  // 5) sequential scan, one workgroup per (b,h)
  ssm_scan_k<<<BATCH * NH, 256, 0, stream>>>(convo, dtb, dAb, Dv, yss);

  // 6) gate + RMSNorm -> bf16
  gate_norm_k<<<BS, 256, 0, stream>>>(yss, projb, norm_w, ybf);

  // 7) out_proj GEMM: [BS,I] x [HIDDEN,I]^T -> [BS,HIDDEN]
  {
    dim3 grid(HIDDEN / 128, BS / 128);
    gemm_bf16_wmma_k<<<grid, 256, 0, stream>>>(ybf, Woutbf, out,
                                               BS, HIDDEN, II);
  }
}
